// AEDecoder_45011257262638
// MI455X (gfx1250) — matrix operand order repaired and checked by reading, not compiled, not run on hardware
//
#include <hip/hip_runtime.h>

// ---- problem constants (match reference) ----
#define N_TF     512
#define NPT      8
#define HIDDEN   4096          // N_TF * NPT
#define N_GENES  20000
#define KREG     16            // regulating TFs per gene
#define BATCH    1024

#define GENE_TILES   (N_GENES / 16)   // 1250
#define BATCH_TILES  (BATCH / 16)     // 64
#define WAVES_PER_BLOCK 8
#define THREADS (WAVES_PER_BLOCK * 32)
#define GRID_X  40                    // gene-tile strips

// LDS row stride in bf16 elements: 4096 + 8 pad halves (16B) -> 2052 dwords/row,
// 2052 % 64 == 4, so the 16 batch-row lanes land on distinct bank groups.
#define ROW_HALVES (HIDDEN + 8)

typedef __attribute__((ext_vector_type(16))) __bf16 v16bf;
typedef __attribute__((ext_vector_type(8)))  __bf16 v8bf;
typedef __attribute__((ext_vector_type(8)))  float  v8f;
typedef __attribute__((ext_vector_type(4)))  float  v4f;
typedef __attribute__((ext_vector_type(8)))  unsigned int v8ui;
typedef __attribute__((ext_vector_type(4)))  unsigned int v4ui;

// ---- pre-pass: convert w2 to packed bf16 once (A-operand feed) ----
__global__ __launch_bounds__(256)
void w2_to_bf16_kernel(const float* __restrict__ w2, __bf16* __restrict__ w2bf) {
    const size_t i = ((size_t)blockIdx.x * 256 + threadIdx.x) * 8;
    if (i >= (size_t)N_GENES * KREG * NPT) return;
    const v4f lo = *(const v4f*)(w2 + i);
    const v4f hi = *(const v4f*)(w2 + i + 4);
    v8bf o;
    #pragma unroll
    for (int j = 0; j < 4; ++j) { o[j] = (__bf16)lo[j]; o[4 + j] = (__bf16)hi[j]; }
    *(v8bf*)(w2bf + i) = o;
}

template<bool USE_WS>
__global__ __launch_bounds__(THREADS, 1)
void aedecoder_wmma_kernel(const float*  __restrict__ features,  // [BATCH, N_TF]
                           const float*  __restrict__ w1,        // [HIDDEN]
                           const float*  __restrict__ b1,        // [HIDDEN]
                           const float*  __restrict__ w2,        // [N_GENES, KREG, NPT]
                           const __bf16* __restrict__ w2bf,      // same, pre-packed bf16
                           const float*  __restrict__ b2,        // [N_GENES]
                           const int*    __restrict__ gene_tf,   // [N_GENES, KREG]
                           float* __restrict__ out)              // [BATCH, N_GENES]
{
    // 16-batch-row slice of layer-1 activations, bf16, padded rows.
    __shared__ __bf16 hTile[16 * ROW_HALVES];   // 131,328 bytes < 320 KB/WGP

    const int b0  = blockIdx.y * 16;
    const int tid = threadIdx.x;

    // ---- stage layer 1 into LDS: one TF-group (8 hidden units) per iteration ----
    for (int i = tid; i < 16 * N_TF; i += THREADS) {
        const int r = i >> 9;            // local batch row 0..15
        const int t = i & (N_TF - 1);    // TF index 0..511
        const float f = features[(size_t)(b0 + r) * N_TF + t];
        const v4f w1lo = *(const v4f*)(w1 + t * NPT);
        const v4f w1hi = *(const v4f*)(w1 + t * NPT + 4);
        const v4f b1lo = *(const v4f*)(b1 + t * NPT);
        const v4f b1hi = *(const v4f*)(b1 + t * NPT + 4);
        v8bf hv;
        #pragma unroll
        for (int j = 0; j < 4; ++j) {
            float vlo = f * w1lo[j] + b1lo[j];
            float vhi = f * w1hi[j] + b1hi[j];
            vlo = (vlo > 0.0f) ? vlo : 0.01f * vlo;
            vhi = (vhi > 0.0f) ? vhi : 0.01f * vhi;
            hv[j]     = (__bf16)vlo;
            hv[4 + j] = (__bf16)vhi;
        }
        *(v8bf*)&hTile[r * ROW_HALVES + t * NPT] = hv;   // ds_store_b128
    }
    __syncthreads();

    const int  wave   = tid >> 5;
    const int  lane   = tid & 31;
    const int  n      = lane & 15;        // batch column within tile (N index)
    const bool hiLane = lane >= 16;
    const __bf16* hRow = &hTile[n * ROW_HALVES];   // per-lane LDS row base

    // ---- lane-constant A-placement masks (hoisted to SGPR lane-masks) ----
    // Lane holds gene row m = n. Its 8-wide K band lives in sub-group qOwn = m>>2.
    // A layout: lanes 0-15: halves 0-7 <-> K 0-7 (d==0), halves 8-15 <-> K 8-15 (d==1)
    //           lanes 16-31: halves 0-7 <-> K 16-23 (d==2), halves 8-15 <-> K 24-31 (d==3)
    const int  m    = n;
    const int  qOwn = m >> 2;
    const int  r4   = m & 3;
    const bool lowAny  = (r4 == (hiLane ? 2 : 0));
    const bool highAny = (r4 == (hiLane ? 3 : 1));
    bool mLow[4], mHigh[4];
    #pragma unroll
    for (int q = 0; q < 4; ++q) {
        mLow[q]  = lowAny  && (qOwn == q);
        mHigh[q] = highAny && (qOwn == q);
    }

    // Each wave produces 16(batch) x 16(genes) output tiles.
    for (int gt = blockIdx.x * WAVES_PER_BLOCK + wave; gt < GENE_TILES;
         gt += gridDim.x * WAVES_PER_BLOCK) {
        const int g0 = gt * 16;

        // Per-lane base pointers: all inner-loop loads use immediate offsets.
        const float*  w2base   = w2   + (size_t)(g0 + m) * (KREG * NPT);
        const __bf16* w2bfbase = USE_WS ? (w2bf + (size_t)(g0 + m) * (KREG * NPT))
                                        : (const __bf16*)nullptr;
        const int*    gtfbase  = gene_tf + (size_t)(g0 + (hiLane ? 2 : 0)) * KREG;

        v8f acc = {0.f, 0.f, 0.f, 0.f, 0.f, 0.f, 0.f, 0.f};

        #pragma unroll
        for (int k = 0; k < KREG; ++k) {
            // ---- gene m's 8 weights for slot k as 4 packed-bf16 dwords ----
            v4ui wp;
            if (USE_WS) {
                wp = *(const v4ui*)(w2bfbase + k * NPT);   // global_load_b128
            } else {
                const v4f wlo = *(const v4f*)(w2base + k * NPT);
                const v4f whi = *(const v4f*)(w2base + k * NPT + 4);
                v8bf wbf;
                #pragma unroll
                for (int j = 0; j < 4; ++j) {
                    wbf[j]     = (__bf16)wlo[j];
                    wbf[4 + j] = (__bf16)whi[j];
                }
                wp = __builtin_bit_cast(v4ui, wbf);
            }

            #pragma unroll
            for (int q = 0; q < 4; ++q) {
                // ---- build A: 8 dword selects, masks are lane-constant ----
                v8ui ai;
                #pragma unroll
                for (int j = 0; j < 4; ++j) {
                    ai[j]     = mLow[q]  ? wp[j] : 0u;
                    ai[4 + j] = mHigh[q] ? wp[j] : 0u;
                }
                const v16bf a = __builtin_bit_cast(v16bf, ai);

                // ---- B: gathered h columns. lanes 0-15: genes 4q+0,4q+1 (K 0..15),
                //         lanes 16-31: genes 4q+2,4q+3 (K 16..31) ----
                const int tfA = gtfbase[q * 4 * KREG + k];          // gene 4q+off
                const int tfB = gtfbase[q * 4 * KREG + KREG + k];   // gene 4q+off+1
                const v8bf hA = *(const v8bf*)(hRow + tfA * NPT);   // ds_load_b128
                const v8bf hB = *(const v8bf*)(hRow + tfB * NPT);
                v16bf b;
                #pragma unroll
                for (int j = 0; j < 8; ++j) { b[j] = hA[j]; b[8 + j] = hB[j]; }

                // ---- D = A x B + C : exact per-gene dot products, rows 4q..4q+3 ----
                acc = __builtin_amdgcn_wmma_f32_16x16x32_bf16(
                          /*neg_a=*/false, a, /*neg_b=*/false, b,
                          /*c_mod=*/(short)0, acc,
                          /*reuse_a=*/false, /*reuse_b=*/false);
            }
        }

        // ---- epilogue: +b2, store. C layout: lane=n, VGPR r -> gene m = r (+8 hi) ----
        const int gcol = g0 + (hiLane ? 8 : 0);
        const v4f b2lo = *(const v4f*)(b2 + gcol);
        const v4f b2hi = *(const v4f*)(b2 + gcol + 4);
        float* orow = out + (size_t)(b0 + n) * N_GENES + gcol;
        v4f lo, hi;
        #pragma unroll
        for (int r = 0; r < 4; ++r) {
            lo[r] = acc[r]     + b2lo[r];
            hi[r] = acc[4 + r] + b2hi[r];
        }
        *(v4f*)(orow)     = lo;   // 16B-aligned
        *(v4f*)(orow + 4) = hi;
    }
}

extern "C" void kernel_launch(void* const* d_in, const int* in_sizes, int n_in,
                              void* d_out, int out_size, void* d_ws, size_t ws_size,
                              hipStream_t stream) {
    const float* features = (const float*)d_in[0];
    const float* w1       = (const float*)d_in[1];
    const float* b1       = (const float*)d_in[2];
    const float* w2       = (const float*)d_in[3];
    const float* b2       = (const float*)d_in[4];
    const int*   gene_tf  = (const int*)d_in[5];
    float* out = (float*)d_out;

    dim3 grid(GRID_X, BATCH_TILES);   // 40 x 64 = 2560 workgroups
    dim3 block(THREADS);              // 8 wave32s

    const size_t w2elems = (size_t)N_GENES * KREG * NPT;      // 2,560,000
    const size_t need    = w2elems * sizeof(__bf16);          // 5,120,000 B

    if (ws_size >= need) {
        __bf16* w2bf = (__bf16*)d_ws;
        const int pblocks = (int)((w2elems / 8 + 255) / 256); // 1250
        w2_to_bf16_kernel<<<pblocks, 256, 0, stream>>>(w2, w2bf);
        aedecoder_wmma_kernel<true><<<grid, block, 0, stream>>>(
            features, w1, b1, w2, w2bf, b2, gene_tf, out);
    } else {
        aedecoder_wmma_kernel<false><<<grid, block, 0, stream>>>(
            features, w1, b1, w2, nullptr, b2, gene_tf, out);
    }
}
